// Attention2_80771154968812
// MI455X (gfx1250) — matrix operand order repaired and checked by reading, not compiled, not run on hardware
//
#include <hip/hip_runtime.h>
#include <hip/hip_bf16.h>

typedef __bf16  v16bf __attribute__((ext_vector_type(16)));
typedef __bf16  v8bf  __attribute__((ext_vector_type(8)));
typedef float   v8f   __attribute__((ext_vector_type(8)));

#define DEV_INLINE __device__ __forceinline__

DEV_INLINE float cexpf(float z) {
    z = fminf(fmaxf(z, -20.0f), 20.0f);
    return __expf(z);
}

DEV_INLINE v16bf cat16(v8bf lo, v8bf hi) {
    return __builtin_shufflevector(lo, hi,
                                   0, 1, 2, 3, 4, 5, 6, 7,
                                   8, 9, 10, 11, 12, 13, 14, 15);
}

// ---------------------------------------------------------------------------
// Pack / convert kernels (f32 -> bf16 staging for WMMA)
// ---------------------------------------------------------------------------

// x [B,N,W] f32 -> A matrix [M=B*N, K=W] bf16 (row-major, same layout)
__global__ void pack_x_kernel(const float* __restrict__ x, __bf16* __restrict__ Xb, int total) {
    int i = blockIdx.x * blockDim.x + threadIdx.x;
    if (i < total) Xb[i] = (__bf16)x[i];
}

// k1,k2,k3 [H,W] -> Kt [64, W] bf16 : row col = j*16+h (j=0..2), rows 48..63 zero
__global__ void pack_kt_kernel(const float* __restrict__ k1, const float* __restrict__ k2,
                               const float* __restrict__ k3, __bf16* __restrict__ Kt, int W) {
    int i = blockIdx.x * blockDim.x + threadIdx.x;
    if (i >= 64 * W) return;
    int col = i / W, w = i % W;
    float v = 0.0f;
    if (col < 48) {
        int j = col >> 4, h = col & 15;
        const float* k = (j == 0) ? k1 : (j == 1) ? k2 : k3;
        v = k[h * W + w];
    }
    Kt[i] = (__bf16)v;
}

// value_weight [H,W,A] -> Vt [HA=1024 rows, K=W] bf16 : Vt[h*64+a][w] = vw[h][w][a]
__global__ void pack_vt_kernel(const float* __restrict__ vw, __bf16* __restrict__ Vt, int W, int A) {
    int i = blockIdx.x * blockDim.x + threadIdx.x;
    if (i >= 1024 * W) return;
    int ha = i / W, w = i % W;
    int h = ha >> 6, a = ha & 63;
    Vt[i] = (__bf16)vw[((size_t)h * W + w) * A + a];
}

// output_weight [H,W,A] -> Ot [W rows(=Ncol), K=HA] bf16 : Ot[w][h*64+a] = ow[h][w][a]
__global__ void pack_ot_kernel(const float* __restrict__ ow, __bf16* __restrict__ Ot, int W, int A) {
    int i = blockIdx.x * blockDim.x + threadIdx.x;
    if (i >= W * 1024) return;
    int w = i / 1024, ha = i % 1024;
    int h = ha >> 6, a = ha & 63;
    Ot[i] = (__bf16)ow[((size_t)h * W + w) * A + a];
}

// ---------------------------------------------------------------------------
// Sinusoid tables p1,p2 [H,N]
// ---------------------------------------------------------------------------
__global__ void phase_kernel(const float* __restrict__ a1, const float* __restrict__ a2,
                             const float* __restrict__ b1, const float* __restrict__ b2,
                             const float* __restrict__ c,
                             float* __restrict__ p1, float* __restrict__ p2,
                             int H, int N, int P) {
    int idx = blockIdx.x * blockDim.x + threadIdx.x;
    if (idx >= H * N) return;
    int h = idx / N, n = idx % N;
    float t = (float)n / (float)(N - 1);
    float s1 = 0.0f, s2 = 0.0f;
    for (int p = 0; p < P; ++p) {
        float cc = c[h * P + p];
        s1 += __sinf(__fmaf_rn(a1[h * P + p], t, b1[h * P + p])) * cc;
        s2 += __sinf(__fmaf_rn(a2[h * P + p], t, b2[h * P + p])) * cc;
    }
    p1[idx] = s1;
    p2[idx] = s2;
}

// ---------------------------------------------------------------------------
// WMMA bf16 GEMM: C[M,Ncol] OutT = A[M,K] * Bt[Ncol,K]^T
// One wave computes a 16x64 strip (4 accumulators), k-stepped by 32 with a
// one-iteration register double buffer so the next k-step's global loads are
// in flight while the current step's 4 WMMAs execute (relaxes s_wait_loadcnt
// from 0 to the in-flight-load count).
// A fragment layout per CDNA5 ISA (16-bit A 16x32):
//   lane row = lane%16 ; contiguous K chunks at (lane/16)*8 and 16+(lane/16)*8
// B fragment (32x16) from transposed storage: col = lane%16,
//   contiguous K chunks at (lane/16)*8 and 16+(lane/16)*8 (same as A).
// ---------------------------------------------------------------------------
template <typename OutT>
__global__ void gemm_bf16_kernel(const __bf16* __restrict__ A,
                                 const __bf16* __restrict__ Bt,
                                 OutT* __restrict__ C,
                                 int M, int Ncol, int K) {
    const int NT = 4;                         // 4 column tiles per wave -> 16x64
    const int lane = threadIdx.x & 31;
    const int wavesPerBlock = blockDim.x >> 5;
    const int wave = blockIdx.x * wavesPerBlock + (threadIdx.x >> 5);
    const int tilesN = Ncol / (16 * NT);
    const int tilesM = M / 16;
    if (wave >= tilesM * tilesN) return;
    const int tm = wave / tilesN;
    const int tn = wave % tilesN;
    const int row0 = tm * 16;
    const int col0 = tn * 16 * NT;

    const int arow = lane & 15;
    const int ksubA = (lane >> 4) * 8;        // 0 or 8
    const int bcol = lane & 15;
    const int ksubB = (lane >> 4) * 8;        // 0 or 8

    const __bf16* Arow = A + (size_t)(row0 + arow) * K + ksubA;
    const __bf16* Bbase[NT];
#pragma unroll
    for (int t = 0; t < NT; ++t)
        Bbase[t] = Bt + (size_t)(col0 + t * 16 + bcol) * K + ksubB;

    const v8f vzero = {0.f, 0.f, 0.f, 0.f, 0.f, 0.f, 0.f, 0.f};
    v8f acc[NT];
#pragma unroll
    for (int t = 0; t < NT; ++t) acc[t] = vzero;

    // Preload k = 0
    v8bf aLo = *(const v8bf*)(Arow);
    v8bf aHi = *(const v8bf*)(Arow + 16);
    v8bf bLo[NT], bHi[NT];
#pragma unroll
    for (int t = 0; t < NT; ++t) {
        bLo[t] = *(const v8bf*)(Bbase[t]);
        bHi[t] = *(const v8bf*)(Bbase[t] + 16);
    }

    for (int k0 = 32; k0 < K; k0 += 32) {
        // Stash current fragments
        v16bf afrag = cat16(aLo, aHi);
        v16bf bfrag[NT];
#pragma unroll
        for (int t = 0; t < NT; ++t) bfrag[t] = cat16(bLo[t], bHi[t]);
        // Issue next k-step's loads (double buffer)
        aLo = *(const v8bf*)(Arow + k0);
        aHi = *(const v8bf*)(Arow + k0 + 16);
#pragma unroll
        for (int t = 0; t < NT; ++t) {
            bLo[t] = *(const v8bf*)(Bbase[t] + k0);
            bHi[t] = *(const v8bf*)(Bbase[t] + k0 + 16);
        }
        // Compute on the stashed fragments
#pragma unroll
        for (int t = 0; t < NT; ++t) {
            acc[t] = __builtin_amdgcn_wmma_f32_16x16x32_bf16(
                false, afrag, false, bfrag[t], (short)0, acc[t], false, false);
        }
    }
    // Epilogue k-step
    {
        v16bf afrag = cat16(aLo, aHi);
#pragma unroll
        for (int t = 0; t < NT; ++t) {
            v16bf bfrag = cat16(bLo[t], bHi[t]);
            acc[t] = __builtin_amdgcn_wmma_f32_16x16x32_bf16(
                false, afrag, false, bfrag, (short)0, acc[t], false, false);
        }
    }

    // C/D layout: VGPR r -> (M = r + (lane>=16 ? 8 : 0), N = lane&15)
    const int srow = row0 + ((lane >> 4) * 8);
    const int scol = lane & 15;
#pragma unroll
    for (int t = 0; t < NT; ++t) {
#pragma unroll
        for (int r = 0; r < 8; ++r) {
            C[(size_t)(srow + r) * Ncol + col0 + t * 16 + scol] = (OutT)acc[t][r];
        }
    }
}

// ---------------------------------------------------------------------------
// Gating: crossE/diagE/gE [B,H,N] from scores S [M,64] + p1/p2 [H,N]
// S col layout: j*16+h with j=0:cross 1:diag 2:extra
// ---------------------------------------------------------------------------
__global__ void gate_kernel(const float* __restrict__ S, const float* __restrict__ p1,
                            const float* __restrict__ p2,
                            float* __restrict__ crossE, float* __restrict__ diagE,
                            float* __restrict__ gE, int B, int H, int N) {
    int idx = blockIdx.x * blockDim.x + threadIdx.x;   // b*H*N + h*N + n
    if (idx >= B * H * N) return;
    int n = idx % N;
    int bh = idx / N;
    int h = bh % H, b = bh / H;
    size_t row = (size_t)b * N + n;
    float cr = S[row * 64 + h] + p1[h * N + n];
    float dg = S[row * 64 + 16 + h];
    float ex = S[row * 64 + 32 + h];
    crossE[idx] = cexpf(cr);
    diagE[idx]  = cexpf(dg);
    gE[idx]     = cexpf(p2[h * N + n]) * cexpf(ex);
}

// ---------------------------------------------------------------------------
// Scan pass 1: per-(b,h,chunk) sums of crossE and crossE*values
// Vp (values) layout: [b*N+n][h*64+a] bf16.  block = (b*H+h)*Cn + ch, 64 threads (a)
// ---------------------------------------------------------------------------
__global__ void chunk_sum_kernel(const float* __restrict__ crossE, const __bf16* __restrict__ Vp,
                                 float* __restrict__ sumC, float* __restrict__ sumCV,
                                 int B, int H, int N, int Cn, int L) {
    int blk = blockIdx.x;
    int ch = blk % Cn;
    int bh = blk / Cn;
    int h = bh % H, b = bh / H;
    int a = threadIdx.x;
    const float* ce = crossE + (size_t)bh * N + ch * L;
    const __bf16* vp = Vp + ((size_t)b * N + (size_t)ch * L) * 1024 + h * 64;
    float sc = 0.0f, scv = 0.0f;
    for (int i = 0; i < L; ++i) {
        float c0 = ce[i];
        sc += c0;
        scv = __fmaf_rn(c0, (float)vp[(size_t)i * 1024 + a], scv);
    }
    sumCV[(size_t)blk * 64 + a] = scv;
    if (a == 0) sumC[blk] = sc;
}

// Scan pass 2: exclusive scan over Cn chunks per (b,h); in-place
__global__ void chunk_scan_kernel(float* __restrict__ sumC, float* __restrict__ sumCV, int Cn) {
    int bh = blockIdx.x;
    int a = threadIdx.x;
    float runV = 0.0f, runC = 0.0f;
    for (int ch = 0; ch < Cn; ++ch) {
        size_t idx = ((size_t)(bh * Cn + ch)) * 64 + a;
        float v = sumCV[idx];
        sumCV[idx] = runV;
        runV += v;
        if (a == 0) {
            float s = sumC[bh * Cn + ch];
            sumC[bh * Cn + ch] = runC;
            runC += s;
        }
    }
}

// Scan pass 3: inclusive cumsum with chunk offset, gate, divide, pack to bf16
__global__ void scan_final_kernel(const float* __restrict__ crossE, const float* __restrict__ diagE,
                                  const float* __restrict__ gE, const __bf16* __restrict__ Vp,
                                  const float* __restrict__ sumC, const float* __restrict__ sumCV,
                                  __bf16* __restrict__ outP,
                                  int B, int H, int N, int Cn, int L) {
    int blk = blockIdx.x;
    int ch = blk % Cn;
    int bh = blk / Cn;
    int h = bh % H, b = bh / H;
    int a = threadIdx.x;
    float runV = sumCV[(size_t)blk * 64 + a];
    float runC = sumC[blk];
    const float* ce = crossE + (size_t)bh * N + ch * L;
    const float* de = diagE + (size_t)bh * N + ch * L;
    const float* ge = gE + (size_t)bh * N + ch * L;
    const __bf16* vp = Vp + ((size_t)b * N + (size_t)ch * L) * 1024 + h * 64;
    __bf16* op = outP + ((size_t)b * N + (size_t)ch * L) * 1024 + h * 64;
    for (int i = 0; i < L; ++i) {
        float c0 = ce[i], d0 = de[i], g0 = ge[i];
        float v = (float)vp[(size_t)i * 1024 + a];
        runC += c0;
        runV = __fmaf_rn(c0, v, runV);
        float num = __fmaf_rn(runV, g0, v * d0);
        float den = __fmaf_rn(runC, g0, d0);
        op[(size_t)i * 1024 + a] = (__bf16)(num / den);
    }
}

// ---------------------------------------------------------------------------
// Launch
// ---------------------------------------------------------------------------
extern "C" void kernel_launch(void* const* d_in, const int* in_sizes, int n_in,
                              void* d_out, int out_size, void* d_ws, size_t ws_size,
                              hipStream_t stream) {
    (void)in_sizes; (void)n_in; (void)out_size; (void)ws_size;
    const int B = 4, N = 4096, W = 1024, H = 16, A = 64, P = 64;
    const int M = B * N;              // 16384
    const int HA = H * A;             // 1024
    const int Cn = 32, L = N / Cn;    // 32 chunks of 128

    const float* x  = (const float*)d_in[0];
    const float* k1 = (const float*)d_in[1];
    const float* k2 = (const float*)d_in[2];
    const float* k3 = (const float*)d_in[3];
    const float* a1 = (const float*)d_in[4];
    const float* a2 = (const float*)d_in[5];
    const float* b1 = (const float*)d_in[6];
    const float* b2 = (const float*)d_in[7];
    const float* c  = (const float*)d_in[8];
    const float* vw = (const float*)d_in[9];
    const float* ow = (const float*)d_in[10];
    float* out = (float*)d_out;

    size_t off = 0;
    auto alloc = [&](size_t bytes) -> void* {
        off = (off + 255) & ~(size_t)255;
        void* p = (char*)d_ws + off;
        off += bytes;
        return p;
    };
    __bf16* Xb     = (__bf16*)alloc((size_t)M * W * 2);      // 32 MB
    __bf16* Kt     = (__bf16*)alloc((size_t)64 * W * 2);     // 128 KB
    __bf16* Vt     = (__bf16*)alloc((size_t)HA * W * 2);     // 2 MB
    __bf16* Ot     = (__bf16*)alloc((size_t)W * HA * 2);     // 2 MB
    float*  Sc     = (float*)alloc((size_t)M * 64 * 4);      // 4 MB
    __bf16* Vp     = (__bf16*)alloc((size_t)M * HA * 2);     // 32 MB
    float*  p1     = (float*)alloc((size_t)H * N * 4);
    float*  p2     = (float*)alloc((size_t)H * N * 4);
    float*  crossE = (float*)alloc((size_t)B * H * N * 4);
    float*  diagE  = (float*)alloc((size_t)B * H * N * 4);
    float*  gE     = (float*)alloc((size_t)B * H * N * 4);
    float*  sumC   = (float*)alloc((size_t)B * H * Cn * 4);
    float*  sumCV  = (float*)alloc((size_t)B * H * Cn * 64 * 4);
    __bf16* outP   = (__bf16*)alloc((size_t)M * HA * 2);     // 32 MB

    const int TB = 256;
    // Stage bf16 operands
    pack_x_kernel<<<(M * W + TB - 1) / TB, TB, 0, stream>>>(x, Xb, M * W);
    pack_kt_kernel<<<(64 * W + TB - 1) / TB, TB, 0, stream>>>(k1, k2, k3, Kt, W);
    pack_vt_kernel<<<(HA * W + TB - 1) / TB, TB, 0, stream>>>(vw, Vt, W, A);
    pack_ot_kernel<<<(W * HA + TB - 1) / TB, TB, 0, stream>>>(ow, Ot, W, A);
    phase_kernel<<<(H * N + TB - 1) / TB, TB, 0, stream>>>(a1, a2, b1, b2, c, p1, p2, H, N, P);

    // Scores GEMM: [M,1024] x [1024,64] -> Sc f32 ; waves = (M/16)*(64/64) = 1024
    {
        int waves = (M / 16) * (64 / 64);
        gemm_bf16_kernel<float><<<(waves + 7) / 8, TB, 0, stream>>>(Xb, Kt, Sc, M, 64, W);
    }
    gate_kernel<<<(B * H * N + TB - 1) / TB, TB, 0, stream>>>(Sc, p1, p2, crossE, diagE, gE, B, H, N);

    // Values GEMM: [M,1024] x [1024,1024] -> Vp bf16 ; waves = 1024*16 = 16384
    {
        int waves = (M / 16) * (HA / 64);
        gemm_bf16_kernel<__bf16><<<(waves + 7) / 8, TB, 0, stream>>>(Xb, Vt, Vp, M, HA, W);
    }

    // Chunked scan
    chunk_sum_kernel<<<B * H * Cn, 64, 0, stream>>>(crossE, Vp, sumC, sumCV, B, H, N, Cn, L);
    chunk_scan_kernel<<<B * H, 64, 0, stream>>>(sumC, sumCV, Cn);
    scan_final_kernel<<<B * H * Cn, 64, 0, stream>>>(crossE, diagE, gE, Vp, sumC, sumCV, outP,
                                                     B, H, N, Cn, L);

    // Output GEMM: [M,1024] x [1024,1024] -> out f32 [B,N,W]
    {
        int waves = (M / 16) * (W / 64);
        gemm_bf16_kernel<float><<<(waves + 7) / 8, TB, 0, stream>>>(outP, Ot, out, M, W, HA);
    }
}
